// TokenRoutedBlock_27676769255890
// MI455X (gfx1250) — matrix operand order repaired and checked by reading, not compiled, not run on hardware
//
#include <hip/hip_runtime.h>

// ---------------- problem constants ----------------
#define Bc   2
#define Tc   2048
#define Dc   2048
#define Hc   16
#define HDc  128
#define Kc   4
#define FFc  2048
#define Mc   (Bc * Tc)        // 4096 tokens
#define KKc  (Kc * Dc)        // 8192 reduction dim for mixed linears

typedef __attribute__((ext_vector_type(16))) __bf16 v16bf;
typedef __attribute__((ext_vector_type(8)))  float  v8f;

union FragBF { v16bf v; uint4 q[2]; };

__device__ __forceinline__ v8f wmma_bf16(v16bf a, v16bf b, v8f c) {
    // D = A(16x32 bf16) * B(32x16 bf16) + C(16x16 f32)
    return __builtin_amdgcn_wmma_f32_16x16x32_bf16(
        /*neg_a=*/false, a, /*neg_b=*/false, b,
        /*c_mod=*/(short)0, c, /*reuse_a=*/false, /*reuse_b=*/false);
}

// CDNA5 async global->LDS copy, 16 bytes per lane, tracked by ASYNCcnt.
// GVS addressing: mem = SGPR base + signed 32-bit per-lane byte offset.
// vdst holds the per-lane LDS byte address (low 32 bits of flat LDS address).
__device__ __forceinline__ void async_copy_b128(uint32_t lds_addr,
                                                int byte_off,
                                                const void* base) {
    asm volatile("global_load_async_to_lds_b128 %0, %1, %2"
                 :: "v"(lds_addr), "v"(byte_off), "s"(base)
                 : "memory");
}
__device__ __forceinline__ void wait_async0() {
    asm volatile("s_wait_asynccnt 0" ::: "memory");
}
__device__ __forceinline__ uint32_t lds_addr_of(const void* p) {
    return (uint32_t)(uintptr_t)p;   // LDS aperture: addr[31:0] is the LDS offset
}

// ---------------------------------------------------------------------------
// Weight conversion: bases (K, O, Din) f32  ->  W' (O, K*Din) bf16, k-major
// ---------------------------------------------------------------------------
__global__ __launch_bounds__(256) void convert_weights_kernel(
    const float* __restrict__ src, __bf16* __restrict__ dst, int O, int Din)
{
    const size_t n4 = (size_t)O * Kc * Din / 4;
    const size_t idx = (size_t)blockIdx.x * blockDim.x + threadIdx.x;
    if (idx >= n4) return;
    const size_t e = idx * 4;
    const int o = (int)(e / ((size_t)Kc * Din));
    const size_t r = e % ((size_t)Kc * Din);
    const int k = (int)(r / Din);
    const int d = (int)(r % Din);
    const float4 v = *(const float4*)(src + ((size_t)k * O + o) * Din + d);
    dst[e + 0] = (__bf16)v.x;
    dst[e + 1] = (__bf16)v.y;
    dst[e + 2] = (__bf16)v.z;
    dst[e + 3] = (__bf16)v.w;
}

// ---------------------------------------------------------------------------
// RMSNorm: h = x * rsqrt(mean(x^2)+eps) * norm_w * gamma + beta   (row = token)
// ---------------------------------------------------------------------------
__global__ __launch_bounds__(256) void rmsnorm_kernel(
    const float* __restrict__ x, const float* __restrict__ w,
    const float* __restrict__ gamma, const float* __restrict__ beta,
    float* __restrict__ h)
{
    __shared__ float red[256];
    const int m = blockIdx.x, tid = threadIdx.x;
    const float* xr = x + (size_t)m * Dc;
    float xv[8];
    float ss = 0.f;
#pragma unroll
    for (int e = 0; e < 8; ++e) { xv[e] = xr[tid + e * 256]; ss += xv[e] * xv[e]; }
    red[tid] = ss;
    __syncthreads();
    for (int s = 128; s > 0; s >>= 1) {
        if (tid < s) red[tid] += red[tid + s];
        __syncthreads();
    }
    const float scale = rsqrtf(red[0] / (float)Dc + 1.1920929e-07f);
    float* hr = h + (size_t)m * Dc;
#pragma unroll
    for (int e = 0; e < 8; ++e) {
        const int d = tid + e * 256;
        hr[d] = xv[e] * scale * w[d] * gamma[d] + beta[d];
    }
}

// ---------------------------------------------------------------------------
// Routing + expand: logits_k = <x, route[li,k]>/sqrt(D) + bias[li,k];
// alpha = softmax(logits); A[m][k*D + d] = bf16(alpha_k * x[d])
// ---------------------------------------------------------------------------
__global__ __launch_bounds__(256) void route_expand_kernel(
    const float* __restrict__ x, const float* __restrict__ route,
    const float* __restrict__ bias, const int* __restrict__ li_p,
    __bf16* __restrict__ Aout, float* __restrict__ alpha_out)
{
    __shared__ float red[256];
    __shared__ float al[Kc];
    const int m = blockIdx.x, tid = threadIdx.x;
    const int li = *li_p;
    const float* xr = x + (size_t)m * Dc;
    float xv[8];
#pragma unroll
    for (int e = 0; e < 8; ++e) xv[e] = xr[tid + e * 256];

    const float* rb = route + (size_t)li * Kc * Dc;
    float part[Kc];
#pragma unroll
    for (int k = 0; k < Kc; ++k) {
        const float* rr = rb + (size_t)k * Dc;
        float p = 0.f;
#pragma unroll
        for (int e = 0; e < 8; ++e) p += xv[e] * rr[tid + e * 256];
        part[k] = p;
    }
    for (int k = 0; k < Kc; ++k) {
        red[tid] = part[k];
        __syncthreads();
        for (int s = 128; s > 0; s >>= 1) {
            if (tid < s) red[tid] += red[tid + s];
            __syncthreads();
        }
        if (tid == 0) al[k] = red[0] * 0.022097086912079608f /* 1/sqrt(2048) */
                              + bias[li * Kc + k];
        __syncthreads();
    }
    if (tid == 0) {
        float mx = al[0];
        for (int k = 1; k < Kc; ++k) mx = fmaxf(mx, al[k]);
        float s = 0.f;
        for (int k = 0; k < Kc; ++k) { al[k] = __expf(al[k] - mx); s += al[k]; }
        const float inv = 1.f / s;
        for (int k = 0; k < Kc; ++k) al[k] *= inv;
    }
    __syncthreads();
    __bf16* Ar = Aout + (size_t)m * KKc;
#pragma unroll
    for (int k = 0; k < Kc; ++k) {
        const float a = al[k];
#pragma unroll
        for (int e = 0; e < 8; ++e) {
            const int d = tid + e * 256;
            Ar[k * Dc + d] = (__bf16)(a * xv[e]);
        }
    }
    if (alpha_out && tid < Kc) alpha_out[(size_t)m * Kc + tid] = al[tid];
}

// ---------------------------------------------------------------------------
// bf16 WMMA GEMM: C[M,N] = A[M,KK] * Bw[N,KK]^T
// Block tile 128x128x32, 8 waves (wave32) as 4x2, wave tile 32x64 = 2x4 WMMA.
// Double-buffered LDS fed by GLOBAL_LOAD_ASYNC_TO_LDS_B128 (ASYNCcnt).
//  MODE 0: store bf16
//  MODE 1: f32 out = res0 + acc                (residual add)
//  MODE 2: f32 out = acc
//  MODE 3: f32 out = silu(res0) * acc          (SwiGLU combine)
//  MODE 4: f32 out = res0 + (res1 + acc - res0) * (*scale_p)   (final blend)
// ---------------------------------------------------------------------------
#define BM 128
#define BN 128
#define BK 32

template <int MODE>
__global__ __launch_bounds__(256) void gemm_bf16_kernel(
    const __bf16* __restrict__ A, const __bf16* __restrict__ Bw,
    int N, int KK, void* __restrict__ outp,
    const float* __restrict__ res0, const float* __restrict__ res1,
    const float* __restrict__ scale_p)
{
    __shared__ __bf16 As[2][BM][BK];
    __shared__ __bf16 Bs[2][BN][BK];

    const int tid = threadIdx.x;
    const int wave = tid >> 5, lane = tid & 31;
    const int lmod = lane & 15, lhalf = lane >> 4;
    const int wm = wave >> 1, wn = wave & 1;
    const int m0 = blockIdx.y * BM, n0 = blockIdx.x * BN;

    // global->LDS chunk assignment: 512 x 16B chunks per tile, 2 per thread
    const int c0 = tid, c1 = tid + 256;
    const int ar0 = c0 >> 2, ac0 = (c0 & 3) * 8;
    const int ar1 = c1 >> 2, ac1 = (c1 & 3) * 8;

    // byte offsets of this thread's chunks inside tile kk=0
    const int gA0 = (int)((((size_t)(m0 + ar0)) * KK + ac0) * 2);
    const int gA1 = (int)((((size_t)(m0 + ar1)) * KK + ac1) * 2);
    const int gB0 = (int)((((size_t)(n0 + ar0)) * KK + ac0) * 2);
    const int gB1 = (int)((((size_t)(n0 + ar1)) * KK + ac1) * 2);

    // LDS byte addresses of this thread's chunks, both buffers
    uint32_t lA0[2], lA1[2], lB0[2], lB1[2];
#pragma unroll
    for (int b = 0; b < 2; ++b) {
        lA0[b] = lds_addr_of(&As[b][ar0][ac0]);
        lA1[b] = lds_addr_of(&As[b][ar1][ac1]);
        lB0[b] = lds_addr_of(&Bs[b][ar0][ac0]);
        lB1[b] = lds_addr_of(&Bs[b][ar1][ac1]);
    }

    v8f acc[2][4];
    const v8f vzero = {0.f, 0.f, 0.f, 0.f, 0.f, 0.f, 0.f, 0.f};
#pragma unroll
    for (int i = 0; i < 2; ++i)
#pragma unroll
        for (int j = 0; j < 4; ++j) acc[i][j] = vzero;

    // async-copy tile 0 into buffer 0
    async_copy_b128(lA0[0], gA0, A);
    async_copy_b128(lA1[0], gA1, A);
    async_copy_b128(lB0[0], gB0, Bw);
    async_copy_b128(lB1[0], gB1, Bw);
    wait_async0();
    __syncthreads();

    const int nk = KK / BK;
    for (int kt = 0; kt < nk; ++kt) {
        const int cur = kt & 1;
        const bool more = (kt + 1) < nk;
        if (more) {
            const int nxt = cur ^ 1;
            const int kb = (kt + 1) * BK * 2;   // byte advance along K
            async_copy_b128(lA0[nxt], gA0 + kb, A);
            async_copy_b128(lA1[nxt], gA1 + kb, A);
            async_copy_b128(lB0[nxt], gB0 + kb, Bw);
            async_copy_b128(lB1[nxt], gB1 + kb, Bw);
            if (kt + 2 < nk) {
                const int kp = (kt + 2) * BK * 2;
                __builtin_prefetch((const char*)A + gA0 + kp, 0, 1);
                __builtin_prefetch((const char*)Bw + gB0 + kp, 0, 1);
            }
        }
        // A fragment (ISA 16-bit A 16x32 layout): lanes<16 hold k 0..7 & 16..23
        FragBF af[2], bf_[4];
#pragma unroll
        for (int i = 0; i < 2; ++i) {
            const int mI = wm * 32 + i * 16 + lmod;
            af[i].q[0] = *(const uint4*)&As[cur][mI][lhalf * 8];
            af[i].q[1] = *(const uint4*)&As[cur][mI][lhalf * 8 + 16];
        }
        // B fragment: same k layout as A: lanes<16 hold k 0..7 & 16..23
#pragma unroll
        for (int j = 0; j < 4; ++j) {
            const int nI = wn * 64 + j * 16 + lmod;
            bf_[j].q[0] = *(const uint4*)&Bs[cur][nI][lhalf * 8];
            bf_[j].q[1] = *(const uint4*)&Bs[cur][nI][lhalf * 8 + 16];
        }
#pragma unroll
        for (int i = 0; i < 2; ++i)
#pragma unroll
            for (int j = 0; j < 4; ++j)
                acc[i][j] = wmma_bf16(af[i].v, bf_[j].v, acc[i][j]);

        if (more) {
            wait_async0();
            __syncthreads();
        }
    }

    // epilogue: C element (vgpr r): row = r + 8*lhalf, col = lmod
#pragma unroll
    for (int i = 0; i < 2; ++i) {
#pragma unroll
        for (int j = 0; j < 4; ++j) {
#pragma unroll
            for (int r = 0; r < 8; ++r) {
                const int row = m0 + wm * 32 + i * 16 + r + lhalf * 8;
                const int col = n0 + wn * 64 + j * 16 + lmod;
                const size_t idx = (size_t)row * N + col;
                const float v = acc[i][j][r];
                if (MODE == 0) {
                    ((__bf16*)outp)[idx] = (__bf16)v;
                } else if (MODE == 1) {
                    ((float*)outp)[idx] = res0[idx] + v;
                } else if (MODE == 2) {
                    ((float*)outp)[idx] = v;
                } else if (MODE == 3) {
                    const float g = res0[idx];
                    ((float*)outp)[idx] = g / (1.f + __expf(-g)) * v;
                } else {
                    const float xv = res0[idx];
                    const float xa = res1[idx];
                    ((float*)outp)[idx] = xv + (xa + v - xv) * scale_p[0];
                }
            }
        }
    }
}

// ---------------------------------------------------------------------------
// Flash attention (causal), bf16 WMMA. One block per (q-tile of 64, b*h).
// Q tile persistent in LDS (async-loaded); K async per tile; V staged through
// VGPRs for the transpose. Online softmax m/l; O in registers.
// ---------------------------------------------------------------------------
#define QT 64
#define KT 32

__global__ __launch_bounds__(256) void attention_kernel(
    const __bf16* __restrict__ qkv,   // Mc x (3*Dc) : [q | k | v] per head slices
    float* __restrict__ attn)         // Mc x Dc
{
    __shared__ __bf16 Qs[QT][HDc];        // 16 KB
    __shared__ __bf16 Ks[KT][HDc];        // 8 KB
    __shared__ __bf16 Vts[HDc][KT];       // 8 KB  (V transposed: [d][n])
    __shared__ float  Ss[QT][KT];         // 8 KB
    __shared__ __bf16 Ps[QT][KT];         // 4 KB
    __shared__ float  mrow[QT], lrow[QT], crow[QT];

    const int tid = threadIdx.x;
    const int lane = tid & 31, wave = tid >> 5;
    const int lmod = lane & 15, lhalf = lane >> 4;
    const int wm = wave >> 1, wn = wave & 1;     // 4 x 2 wave grid
    const int q0 = blockIdx.x * QT;
    const int bh = (int)blockIdx.y;
    const int b = bh / Hc, hh = bh % Hc;
    const size_t rowbase = (size_t)b * Tc;
    const int ldq = 3 * Dc;
    const int qcol = hh * HDc, kcol = Dc + hh * HDc, vcol = 2 * Dc + hh * HDc;

    // async-load Q tile (64x128 bf16 = 1024 x 16B chunks, 4 per thread)
#pragma unroll
    for (int c = 0; c < 4; ++c) {
        const int ch = tid + c * 256;
        const int rw = ch >> 4, of = (ch & 15) * 8;
        async_copy_b128(lds_addr_of(&Qs[rw][of]),
                        (int)(((rowbase + q0 + rw) * ldq + qcol + of) * 2), qkv);
    }
    if (tid < QT) { mrow[tid] = -1e30f; lrow[tid] = 0.f; }

    // this thread's K-tile chunk (2 per thread) and V-tile chunk bases
    const int chk0 = tid, chk1 = tid + 256;
    const int krw0 = chk0 >> 4, kof0 = (chk0 & 15) * 8;
    const int krw1 = chk1 >> 4, kof1 = (chk1 & 15) * 8;
    const uint32_t lK0 = lds_addr_of(&Ks[krw0][kof0]);
    const uint32_t lK1 = lds_addr_of(&Ks[krw1][kof1]);
    const int gK0 = (int)(((rowbase + krw0) * ldq + kcol + kof0) * 2);
    const int gK1 = (int)(((rowbase + krw1) * ldq + kcol + kof1) * 2);

    const v8f vzero = {0.f, 0.f, 0.f, 0.f, 0.f, 0.f, 0.f, 0.f};
    v8f oacc[4];
#pragma unroll
    for (int t = 0; t < 4; ++t) oacc[t] = vzero;

    const int jmax = (q0 + QT - 1) / KT;  // inclusive: covers causal band
    for (int j = 0; j <= jmax; ++j) {
        const int k0 = j * KT;
        const int kadv = k0 * ldq * 2;    // byte advance for this kv tile
        // K tile async (32x128 = 512 chunks, 2 per thread)
        async_copy_b128(lK0, gK0 + kadv, qkv);
        async_copy_b128(lK1, gK1 + kadv, qkv);
        // V tile, staged through VGPRs and stored transposed [d][n]
#pragma unroll
        for (int c = 0; c < 2; ++c) {
            const int ch = tid + c * 256;
            const int rw = ch >> 4, of = (ch & 15) * 8;
            const uint4 v =
                *(const uint4*)(qkv + (rowbase + k0 + rw) * ldq + vcol + of);
            const __bf16* pv = (const __bf16*)&v;
#pragma unroll
            for (int e = 0; e < 8; ++e) Vts[of + e][rw] = pv[e];
        }
        wait_async0();        // K (and, first iteration, Q) resident in LDS
        __syncthreads();

        // S = Q K^T / sqrt(HD) : 64x32, wave tile 16x16, 4 k-steps over HD
        v8f sacc = vzero;
#pragma unroll
        for (int kt = 0; kt < HDc / 32; ++kt) {
            FragBF a, bb;
            const int mI = wm * 16 + lmod;
            a.q[0] = *(const uint4*)&Qs[mI][kt * 32 + lhalf * 8];
            a.q[1] = *(const uint4*)&Qs[mI][kt * 32 + lhalf * 8 + 16];
            const int nI = wn * 16 + lmod;
            bb.q[0] = *(const uint4*)&Ks[nI][kt * 32 + lhalf * 8];
            bb.q[1] = *(const uint4*)&Ks[nI][kt * 32 + lhalf * 8 + 16];
            sacc = wmma_bf16(a.v, bb.v, sacc);
        }
#pragma unroll
        for (int r = 0; r < 8; ++r) {
            const int rw = wm * 16 + r + lhalf * 8;
            const int cl = wn * 16 + lmod;
            Ss[rw][cl] = sacc[r] * 0.08838834764831845f;  // 1/sqrt(128)
        }
        __syncthreads();

        // online softmax per row (threads 0..63), causal mask on diag tiles
        if (tid < QT) {
            const int rw = tid;
            const int grow = q0 + rw;
            int kv = grow - k0 + 1;
            if (kv > KT) kv = KT;
            if (kv < 0) kv = 0;
            float tmax = -1e30f;
            for (int c = 0; c < kv; ++c) tmax = fmaxf(tmax, Ss[rw][c]);
            const float mold = mrow[rw];
            const float mnew = fmaxf(mold, tmax);
            const float corr = __expf(mold - mnew);
            float s = 0.f;
            for (int c = 0; c < KT; ++c) {
                const float p = (c < kv) ? __expf(Ss[rw][c] - mnew) : 0.f;
                Ps[rw][c] = (__bf16)p;
                s += p;
            }
            lrow[rw] = lrow[rw] * corr + s;
            mrow[rw] = mnew;
            crow[rw] = corr;
        }
        __syncthreads();

        // rescale O, then O += P * V : 64x128, wave tile 16x64 (4 tiles), k=32
#pragma unroll
        for (int t = 0; t < 4; ++t)
#pragma unroll
            for (int r = 0; r < 8; ++r) {
                const int rw = wm * 16 + r + lhalf * 8;
                oacc[t][r] *= crow[rw];
            }
        {
            FragBF a;
            const int mI = wm * 16 + lmod;
            a.q[0] = *(const uint4*)&Ps[mI][lhalf * 8];
            a.q[1] = *(const uint4*)&Ps[mI][lhalf * 8 + 16];
#pragma unroll
            for (int t = 0; t < 4; ++t) {
                FragBF bb;
                const int dI = wn * 64 + t * 16 + lmod;
                bb.q[0] = *(const uint4*)&Vts[dI][lhalf * 8];
                bb.q[1] = *(const uint4*)&Vts[dI][lhalf * 8 + 16];
                oacc[t] = wmma_bf16(a.v, bb.v, oacc[t]);
            }
        }
        __syncthreads();
    }

    // final normalize + store
#pragma unroll
    for (int t = 0; t < 4; ++t)
#pragma unroll
        for (int r = 0; r < 8; ++r) {
            const int rw = wm * 16 + r + lhalf * 8;
            const int cl = wn * 64 + t * 16 + lmod;
            attn[(rowbase + q0 + rw) * Dc + hh * HDc + cl] =
                oacc[t][r] / lrow[rw];
        }
}

// ---------------------------------------------------------------------------
// host-side orchestration (graph-capture safe: launches only)
// ---------------------------------------------------------------------------
extern "C" void kernel_launch(void* const* d_in, const int* in_sizes, int n_in,
                              void* d_out, int out_size, void* d_ws, size_t ws_size,
                              hipStream_t stream) {
    (void)in_sizes; (void)n_in; (void)out_size; (void)ws_size;

    const float* x      = (const float*)d_in[0];
    const float* gamma  = (const float*)d_in[1];
    const float* beta   = (const float*)d_in[2];
    const float* iter_s = (const float*)d_in[3];
    const float* qkv_b  = (const float*)d_in[4];
    const float* o_b    = (const float*)d_in[5];
    const float* gate_b = (const float*)d_in[6];
    const float* up_b   = (const float*)d_in[7];
    const float* down_b = (const float*)d_in[8];
    const float* r_qkv  = (const float*)d_in[9];
    const float* r_o    = (const float*)d_in[10];
    const float* r_gate = (const float*)d_in[11];
    const float* r_up   = (const float*)d_in[12];
    const float* r_down = (const float*)d_in[13];
    const float* b_qkv  = (const float*)d_in[14];
    const float* b_o    = (const float*)d_in[15];
    const float* b_gate = (const float*)d_in[16];
    const float* b_up   = (const float*)d_in[17];
    const float* b_down = (const float*)d_in[18];
    const float* n1w    = (const float*)d_in[19];
    const float* n2w    = (const float*)d_in[20];
    const int*   li     = (const int*)d_in[21];

    uint8_t* ws = (uint8_t*)d_ws;
    size_t off = 0;
    auto carve = [&](size_t bytes) {
        void* p = ws + off;
        off = (off + bytes + 255) & ~(size_t)255;
        return p;
    };
    __bf16* Wqkv = (__bf16*)carve((size_t)(3 * Dc) * KKc * 2);
    __bf16* Wo   = (__bf16*)carve((size_t)Dc * KKc * 2);
    __bf16* Wg   = (__bf16*)carve((size_t)FFc * KKc * 2);
    __bf16* Wu   = (__bf16*)carve((size_t)FFc * KKc * 2);
    __bf16* Wd   = (__bf16*)carve((size_t)Dc * KKc * 2);
    __bf16* Abuf = (__bf16*)carve((size_t)Mc * KKc * 2);
    __bf16* qkvb = (__bf16*)carve((size_t)Mc * 3 * Dc * 2);
    float*  hbuf = (float*)carve((size_t)Mc * Dc * 4);
    float*  attn = (float*)carve((size_t)Mc * Dc * 4);
    float*  xatt = (float*)carve((size_t)Mc * Dc * 4);
    float*  gbuf = (float*)carve((size_t)Mc * Dc * 4);
    float*  fbuf = (float*)carve((size_t)Mc * Dc * 4);

    float* out_x     = (float*)d_out;
    float* out_alpha = out_x + (size_t)Mc * Dc;

    // 0) weights -> bf16, k-major relayout
    auto nblk = [](size_t O) {
        return (unsigned)((O * Kc * 2048 / 4 + 255) / 256);
    };
    convert_weights_kernel<<<nblk(3 * Dc), 256, 0, stream>>>(qkv_b, Wqkv, 3 * Dc, Dc);
    convert_weights_kernel<<<nblk(Dc), 256, 0, stream>>>(o_b, Wo, Dc, Dc);
    convert_weights_kernel<<<nblk(FFc), 256, 0, stream>>>(gate_b, Wg, FFc, Dc);
    convert_weights_kernel<<<nblk(FFc), 256, 0, stream>>>(up_b, Wu, FFc, Dc);
    convert_weights_kernel<<<nblk(Dc), 256, 0, stream>>>(down_b, Wd, Dc, FFc);

    // 1) h1 = rmsnorm(x)*gamma+beta ; route qkv (alpha -> d_out tail) ; qkv GEMM
    rmsnorm_kernel<<<Mc, 256, 0, stream>>>(x, n1w, gamma, beta, hbuf);
    route_expand_kernel<<<Mc, 256, 0, stream>>>(hbuf, r_qkv, b_qkv, li, Abuf, out_alpha);
    gemm_bf16_kernel<0><<<dim3((3 * Dc) / BN, Mc / BM), 256, 0, stream>>>(
        Abuf, Wqkv, 3 * Dc, KKc, qkvb, nullptr, nullptr, nullptr);

    // 2) causal flash attention
    attention_kernel<<<dim3(Tc / QT, Bc * Hc), 256, 0, stream>>>(qkvb, attn);

    // 3) o-projection + residual: x_attn = x + mixed_o(attn)
    route_expand_kernel<<<Mc, 256, 0, stream>>>(attn, r_o, b_o, li, Abuf, nullptr);
    gemm_bf16_kernel<1><<<dim3(Dc / BN, Mc / BM), 256, 0, stream>>>(
        Abuf, Wo, Dc, KKc, xatt, x, nullptr, nullptr);

    // 4) h2 = rmsnorm(x_attn); gate & up mixed linears; ffn = silu(gate)*up
    rmsnorm_kernel<<<Mc, 256, 0, stream>>>(xatt, n2w, gamma, beta, hbuf);
    route_expand_kernel<<<Mc, 256, 0, stream>>>(hbuf, r_gate, b_gate, li, Abuf, nullptr);
    gemm_bf16_kernel<2><<<dim3(FFc / BN, Mc / BM), 256, 0, stream>>>(
        Abuf, Wg, FFc, KKc, gbuf, nullptr, nullptr, nullptr);
    route_expand_kernel<<<Mc, 256, 0, stream>>>(hbuf, r_up, b_up, li, Abuf, nullptr);
    gemm_bf16_kernel<3><<<dim3(FFc / BN, Mc / BM), 256, 0, stream>>>(
        Abuf, Wu, FFc, KKc, fbuf, gbuf, nullptr, nullptr);

    // 5) down mixed linear + final: out = x + (x_attn + down - x) * iter_scale
    route_expand_kernel<<<Mc, 256, 0, stream>>>(fbuf, r_down, b_down, li, Abuf, nullptr);
    gemm_bf16_kernel<4><<<dim3(Dc / BN, Mc / BM), 256, 0, stream>>>(
        Abuf, Wd, Dc, KKc, out_x, x, xatt, iter_s);
}
